// FusedSequenceParallelAttn_68607807586522
// MI455X (gfx1250) — matrix operand order repaired and checked by reading, not compiled, not run on hardware
//
#include <hip/hip_runtime.h>
#include <stdint.h>

#define WORLD   8
#define RANKID  3
#define QHEADS  32
#define KVHEADS 8
#define HD      128
#define SEQ     8192
#define LQ      1024
#define CHUNKSZ 512

typedef __attribute__((ext_vector_type(16))) __bf16          v16bf;
typedef __attribute__((ext_vector_type(8)))  float           v8f;
typedef __attribute__((ext_vector_type(16))) unsigned short  v16us;
typedef __attribute__((ext_vector_type(8)))  unsigned int    v8ui;

__device__ __forceinline__ unsigned short f2bf(float f) {
  unsigned u = __float_as_uint(f);
  u += 0x7fffu + ((u >> 16) & 1u);          // round-to-nearest-even
  return (unsigned short)(u >> 16);
}

// 1/sqrt(128) * log2(e), folded into Q at prep time (softmax done in log2 domain)
#define QK_SCALE 0.12751989454107446f

// ---------------- preprocessing ----------------

// q_bf16[hq][q][d] <- q[q][hq][d] * QK_SCALE   (already in Q^T B-fragment order)
__global__ void prep_q_kernel(const float* __restrict__ q, unsigned short* __restrict__ qb) {
  int i = blockIdx.x * blockDim.x + threadIdx.x;   // QHEADS*LQ*HD
  int d = i % HD;
  int t = i / HD;
  int qi = t % LQ;
  int h  = t / LQ;
  qb[i] = f2bf(q[(qi * QHEADS + h) * HD + d] * QK_SCALE);
}

// K pre-swizzled into WMMA A-fragment order:
// kswz[hk][kt:512][c:4][lane:32][e:16], one 32-byte load per lane per (tile, chunk).
// A k order used here (not the ISA table order): lane<16 holds K(dims) 0..15, lane>=16 holds 16..31 (contiguous, matches Q^T B).
__global__ void prep_k_kernel(const float* __restrict__ k, unsigned short* __restrict__ kswz) {
  int i = blockIdx.x * blockDim.x + threadIdx.x;   // KVHEADS*SEQ*HD
  int e    = i & 15;
  int lane = (i >> 4) & 31;
  int c    = (i >> 9) & 3;
  int kt   = (i >> 11) & 511;
  int hk   = i >> 20;
  int key  = kt * 16 + (lane & 15);
  int d    = c * 32 + ((lane & 16) ? 16 : 0) + e;
  kswz[i] = f2bf(k[(key * KVHEADS + hk) * HD + d]);
}

// V^T pre-swizzled into WMMA A-fragment order (keys are the WMMA K-dim):
// vswz[hk][kg:256][dt:8][lane:32][e:16]
__global__ void prep_v_kernel(const float* __restrict__ v, unsigned short* __restrict__ vswz) {
  int i = blockIdx.x * blockDim.x + threadIdx.x;   // KVHEADS*SEQ*HD
  int e    = i & 15;
  int lane = (i >> 4) & 31;
  int dt   = (i >> 9) & 7;
  int kg   = (i >> 12) & 255;
  int hk   = i >> 20;
  int dim  = dt * 16 + (lane & 15);
  int key  = kg * 32 + ((lane & 16) ? 16 : 0) + e;
  vswz[i] = f2bf(v[(key * KVHEADS + hk) * HD + dim]);
}

// ---------------- flash attention, S^T = K*Q^T / O^T = V^T*P^T formulation ----------------

__global__ __launch_bounds__(256) void zz_attn_kernel(
    const unsigned short* __restrict__ qb,
    const unsigned short* __restrict__ kswz,
    const unsigned short* __restrict__ vswz,
    float* __restrict__ out)
{
  const int lane = threadIdx.x & 31;
  // wave-uniform ids forced into SGPRs: scalar skip branches, SGPR base addressing
  const int wave = __builtin_amdgcn_readfirstlane(threadIdx.x >> 5);
  const int gw   = __builtin_amdgcn_readfirstlane(blockIdx.x) * 8 + wave; // 0..2047
  const int qt   = gw & 63;                 // q tile (16 rows)
  const int hq   = gw >> 6;                 // q head; 8 waves/block share hq -> shared KV head
  const int hk   = hq >> 2;                 // GQA group 4
  const bool lo  = lane < 16;
  const int  l15 = lane & 15;

  // Q^T B-fragments (dims = WMMA K-dim): lane = q column, 32B contiguous per chunk
  const int qrow = qt * 16 + l15;
  const unsigned short* qptr = qb + (hq * LQ + qrow) * HD + (lo ? 0 : 16);
  v16bf qf[4];
#pragma unroll
  for (int c = 0; c < 4; ++c)
    qf[c] = __builtin_bit_cast(v16bf, *(const v16us*)(qptr + c * 32));

  v8f o[8];
  const v8f vzero = {};
#pragma unroll
  for (int t = 0; t < 8; ++t) o[t] = vzero;
  float m = -1e30f, lsum = 0.f;

  // global zig-zag position of this rank's q rows (16-tile never straddles a 512 chunk)
  const int qi0 = qt * 16;
  const int qp0 = (qi0 < CHUNKSZ) ? (RANKID * CHUNKSZ + qi0)
                                  : ((2 * WORLD - 1 - RANKID) * CHUNKSZ + qi0 - CHUNKSZ);
  const int qp  = qp0 + l15;                // per-lane (q column) position

  const unsigned short* kh = kswz + (size_t)hk * SEQ * HD;   // SGPR base
  const unsigned short* vh = vswz + (size_t)hk * SEQ * HD;   // SGPR base
  const int lofs = lane * 16;

  for (int kbi = 0; kbi < SEQ; kbi += 32) {
    // global positions of gathered keys kbi..kbi+31 (contiguous within a 512 chunk) - all SALU
    const int r  = kbi >> 10;
    const int j  = kbi & 1023;
    const int kp0 = (j < CHUNKSZ) ? (r * CHUNKSZ + j)
                                  : ((2 * WORLD - 1 - r) * CHUNKSZ + j - CHUNKSZ);
    if (qp0 + 15 < kp0) continue;                 // scalar skip: group fully above the diagonal
    const bool need_mask = (qp0 < kp0 + 31);      // diagonal group

    const int kt = kbi >> 4;                      // 16-key tile index
    const int kg = kbi >> 5;                      // 32-key group index

    if (kbi + 32 < SEQ) {                         // keep a prefetch stream ahead (L2-resident data)
      __builtin_prefetch(kh + ((kt + 2) << 11) + lane * 32, 0, 0);
      __builtin_prefetch(vh + (((kg + 1) << 3) << 9) + lane * 32, 0, 0);
    }

    // ---- S^T = K * Q^T : two 16-key tiles, 4 dim-chunks of 32, single 32-byte A loads ----
    v8f s0 = vzero, s1 = vzero;
#pragma unroll
    for (int c = 0; c < 4; ++c) {
      v16bf A0 = __builtin_bit_cast(v16bf,
                   *(const v16us*)(kh + (((kt * 4 + c) << 9) + lofs)));
      s0 = __builtin_amdgcn_wmma_f32_16x16x32_bf16(false, A0, false, qf[c], (short)0, s0, false, false);
      v16bf A1 = __builtin_bit_cast(v16bf,
                   *(const v16us*)(kh + ((((kt + 1) * 4 + c) << 9) + lofs)));
      s1 = __builtin_amdgcn_wmma_f32_16x16x32_bf16(false, A1, false, qf[c], (short)0, s1, false, false);
    }

    // zig-zag causal mask (scale already folded into Q; lane = q column, reg = key row)
    if (need_mask) {
#pragma unroll
      for (int i = 0; i < 8; ++i) {
        const int krow = lo ? i : 8 + i;
        if (qp < kp0 + krow)      s0[i] = -1e30f;
        if (qp < kp0 + 16 + krow) s1[i] = -1e30f;
      }
    }

    // ---- online softmax: in-lane reduce over 16 key regs + one xor-16 exchange ----
    float tmax = s0[0];
#pragma unroll
    for (int i = 1; i < 8; ++i) tmax = fmaxf(tmax, s0[i]);
#pragma unroll
    for (int i = 0; i < 8; ++i) tmax = fmaxf(tmax, s1[i]);
    tmax = fmaxf(tmax, __shfl_xor(tmax, 16, 32));
    const float nm    = fmaxf(m, tmax);
    const float alpha = __builtin_exp2f(m - nm);
    float psum = 0.f;
#pragma unroll
    for (int i = 0; i < 8; ++i) {
      s0[i] = __builtin_exp2f(s0[i] - nm); psum += s0[i];
      s1[i] = __builtin_exp2f(s1[i] - nm); psum += s1[i];
    }
    psum += __shfl_xor(psum, 16, 32);
    lsum = lsum * alpha + psum;
    m = nm;
#pragma unroll
    for (int t = 0; t < 8; ++t)
#pragma unroll
      for (int i = 0; i < 8; ++i) o[t][i] *= alpha;

    // ---- pack P^T to bf16 FIRST, then exchange only 8 dwords across the xor-16 halves ----
    unsigned pk0[4], pk1[4];
#pragma unroll
    for (int jj = 0; jj < 4; ++jj) {
      pk0[jj] = (unsigned)f2bf(s0[2 * jj]) | ((unsigned)f2bf(s0[2 * jj + 1]) << 16);
      pk1[jj] = (unsigned)f2bf(s1[2 * jj]) | ((unsigned)f2bf(s1[2 * jj + 1]) << 16);
    }
    v8ui bp;
#pragma unroll
    for (int jj = 0; jj < 4; ++jj) {
      const unsigned o0 = __shfl_xor(pk0[jj], 16, 32);   // partner keys +8..15 (for lo)
      const unsigned o1 = __shfl_xor(pk1[jj], 16, 32);   // partner keys +16..23 (for hi)
      bp[jj]     = lo ? pk0[jj] : o1;
      bp[4 + jj] = lo ? o0 : pk1[jj];
    }
    const v16bf B = __builtin_bit_cast(v16bf, bp);

    // ---- O^T += V^T * P^T : 8 dim-tiles, single 32-byte A loads from swizzled V ----
#pragma unroll
    for (int t = 0; t < 8; ++t) {
      v16bf A = __builtin_bit_cast(v16bf,
                  *(const v16us*)(vh + ((((kg << 3) + t) << 9) + lofs)));
      o[t] = __builtin_amdgcn_wmma_f32_16x16x32_bf16(false, A, false, B, (short)0, o[t], false, false);
    }
  }

  // ---- epilogue: out[q][hq][d] = O^T[d][q] / l[q] ----
  const float inv = 1.0f / lsum;
  float* obase = out + (qrow * QHEADS + hq) * HD + (lo ? 0 : 8);
#pragma unroll
  for (int t = 0; t < 8; ++t)
#pragma unroll
    for (int i = 0; i < 8; ++i)
      obase[t * 16 + i] = o[t][i] * inv;
}

extern "C" void kernel_launch(void* const* d_in, const int* in_sizes, int n_in,
                              void* d_out, int out_size, void* d_ws, size_t ws_size,
                              hipStream_t stream) {
  const float* q = (const float*)d_in[0];
  const float* k = (const float*)d_in[1];
  const float* v = (const float*)d_in[2];
  float* out = (float*)d_out;

  // workspace: Q bf16 (8MB) | K swizzled bf16 (16MB) | V^T swizzled bf16 (16MB) => 40MB
  unsigned short* qb   = (unsigned short*)d_ws;
  unsigned short* kswz = qb   + (size_t)QHEADS * LQ * HD;
  unsigned short* vswz = kswz + (size_t)KVHEADS * SEQ * HD;

  prep_q_kernel<<<(QHEADS * LQ * HD) / 256, 256, 0, stream>>>(q, qb);
  prep_k_kernel<<<(KVHEADS * SEQ * HD) / 256, 256, 0, stream>>>(k, kswz);
  prep_v_kernel<<<(KVHEADS * SEQ * HD) / 256, 256, 0, stream>>>(v, vswz);

  // 2048 waves = 64 q-tiles x 32 heads, 8 waves per block (block shares one KV head for L2 reuse)
  zz_attn_kernel<<<256, 256, 0, stream>>>(qb, kswz, vswz, out);
}
